// SegmentedAttention_CSWIN_bi_20873541058712
// MI455X (gfx1250) — matrix operand order repaired and checked by reading, not compiled, not run on hardware
//
#include <hip/hip_runtime.h>

// ---------------------------------------------------------------------------
// Types for CDNA5 WMMA (wave32): v_wmma_f32_16x16x32_bf16
// ---------------------------------------------------------------------------
typedef __bf16 bf16_t;
typedef __attribute__((ext_vector_type(16))) __bf16        v16bf;
typedef __attribute__((ext_vector_type(8)))  float         v8f;
typedef __attribute__((ext_vector_type(4)))  unsigned int  u32x4;
typedef __attribute__((ext_vector_type(4)))  int           v4i32;

#define WMMA_B(a, b, c) \
  __builtin_amdgcn_wmma_f32_16x16x32_bf16(false, (a), false, (b), (short)0, (c), false, false)

#if __has_builtin(__builtin_amdgcn_global_load_async_to_lds_b128)
#define ASYNC_LDS 1
#else
#define ASYNC_LDS 0
#endif

union F16U { u32x4 q[2]; v16bf v; bf16_t h[16]; };

__device__ __forceinline__ v8f zero8() {
  v8f z = {0.f, 0.f, 0.f, 0.f, 0.f, 0.f, 0.f, 0.f};
  return z;
}

// A-operand fragment (16x32 bf16, row-major), per ISA 7.12.2:
// lanes 0-15 row m=lane, halves h0-7 -> K=8g.., h8-15 -> K=16+8g..
__device__ __forceinline__ v16bf ld_a16(const bf16_t* rowbase, int g) {
  F16U u;
  u.q[0] = *(const u32x4*)(rowbase + 8 * g);
  u.q[1] = *(const u32x4*)(rowbase + 16 + 8 * g);
  return u.v;
}
// B-operand fragment (32x16 bf16) stored as B^T rows:
// lane n holds column n, halves h0-7 -> K=8g.., h8-15 -> K=16+8g..
__device__ __forceinline__ v16bf ld_b16(const bf16_t* colbase, int g) {
  F16U u;
  u.q[0] = *(const u32x4*)(colbase + 8 * g);
  u.q[1] = *(const u32x4*)(colbase + 16 + 8 * g);
  return u.v;
}

// ---------------------------------------------------------------------------
// Async staging helpers (GLOBAL_LOAD_ASYNC_TO_LDS_B128, ASYNCcnt) with a
// sync VGPR-roundtrip fallback if the builtin is unavailable.
// Builtin signature (from compiler diagnostics): arg0 = v4i32 addrspace(1)*,
// arg1 = LDS pointer (addrspace(3)), arg2 = imm offset, arg3 = imm cpol.
// ---------------------------------------------------------------------------
struct APanel { u32x4 a0, a1; };

__device__ __forceinline__ void a_issue(const bf16_t* g, bf16_t* l, APanel& r) {
#if ASYNC_LDS
  // copy 32B: two b128 async transfers, imm offset applies to both addresses
  __builtin_amdgcn_global_load_async_to_lds_b128(
      (__attribute__((address_space(1))) v4i32*)(g),
      (__attribute__((address_space(3))) v4i32*)(l), 0, 0);
  __builtin_amdgcn_global_load_async_to_lds_b128(
      (__attribute__((address_space(1))) v4i32*)(g),
      (__attribute__((address_space(3))) v4i32*)(l), 16, 0);
  (void)r;
#else
  const u32x4* s = (const u32x4*)g;
  r.a0 = s[0];
  r.a1 = s[1];
  (void)l;
#endif
}
__device__ __forceinline__ void a_commit(bf16_t* l, const APanel& r) {
#if !ASYNC_LDS
  u32x4* d = (u32x4*)l;
  d[0] = r.a0;
  d[1] = r.a1;
#else
  (void)l; (void)r;
#endif
}
__device__ __forceinline__ void wait_async() {
#if ASYNC_LDS
#if __has_builtin(__builtin_amdgcn_s_wait_asynccnt)
  __builtin_amdgcn_s_wait_asynccnt(0);
#else
  asm volatile("s_wait_asynccnt 0" ::: "memory");
#endif
#endif
}

struct BPanel { F16U u; };
__device__ __forceinline__ void b_load(const bf16_t* src, BPanel& p) {
  const u32x4* s4 = (const u32x4*)src;
  p.u.q[0] = s4[0];
  p.u.q[1] = s4[1];
}
__device__ __forceinline__ void b_store(bf16_t* bTbuf, int bnc, int bk, const BPanel& p) {
#pragma unroll
  for (int i = 0; i < 16; ++i) bTbuf[(bnc + i) * 40 + bk] = p.u.h[i];
}

// ---------------------------------------------------------------------------
// Problem constants (compile-time; dims fixed by the reference)
// ---------------------------------------------------------------------------
#define BATCH   4
#define NSEQ    16384        // H*W = 128*128
#define DIMM    512
#define HEADS   8
#define DH      64
#define SEG     64
#define NWIN    256          // NSEQ / SEG
#define PMTOK   4
#define ROWS    65536        // BATCH * NSEQ
#define LN1E4_OVER_32 0.28782313662425572f   // ln(10000)/32

// ---------------------------------------------------------------------------
// Kernel 1: RMSNorm -> bf16. One wave per row of 512.
// ---------------------------------------------------------------------------
__global__ __launch_bounds__(256) void k_rmsnorm(const float* __restrict__ x,
                                                 const float* __restrict__ gmm,
                                                 bf16_t* __restrict__ xn) {
  const int wave = threadIdx.x >> 5, lane = threadIdx.x & 31;
  const size_t row = (size_t)blockIdx.x * 8 + wave;
  const float* xr = x + row * DIMM;
  float vals[16];
  float ss = 0.f;
#pragma unroll
  for (int i = 0; i < 16; ++i) {
    float v = xr[lane + 32 * i];
    vals[i] = v;
    ss += v * v;
  }
#pragma unroll
  for (int off = 16; off; off >>= 1) ss += __shfl_xor(ss, off, 32);
  const float r = rsqrtf(ss * (1.0f / (float)DIMM) + 1.1920929e-7f);
  bf16_t* o = xn + row * DIMM;
#pragma unroll
  for (int i = 0; i < 16; ++i)
    o[lane + 32 * i] = (bf16_t)(vals[i] * r * gmm[lane + 32 * i]);
}

// ---------------------------------------------------------------------------
// Kernel 2: cast weights to bf16 (Wq 256K, Wkv 512K, Wo 256K elements)
// ---------------------------------------------------------------------------
__global__ __launch_bounds__(256) void k_castw(const float* __restrict__ wq,
                                               const float* __restrict__ wkv,
                                               const float* __restrict__ wo,
                                               bf16_t* __restrict__ oq,
                                               bf16_t* __restrict__ okv,
                                               bf16_t* __restrict__ oo) {
  const int i = blockIdx.x * 256 + threadIdx.x;
  if (i < 262144)       oq[i] = (bf16_t)wq[i];
  else if (i < 786432)  okv[i - 262144] = (bf16_t)wkv[i - 262144];
  else                  oo[i - 786432] = (bf16_t)wo[i - 786432];
}

// ---------------------------------------------------------------------------
// Kernel 3: WMMA GEMM  C[ROWS x Ncols] = A[ROWS x 512] * W[512 x Ncols]
//   mode 0: Q   -> bf16 scattered (b,h,n,d)
//   mode 1: KV  -> c<512: K bf16 (b,h,n,d); c>=512: V fp32 -> d_out orig_v
//   mode 2: O   -> fp32 row-major (row,512)
//   rowXor: ^0x8000 implements the (b+2)%4 batch-half reversal on A rows.
// Block: 256 thr = 8 waves (2x4), tile 128x128, K-steps of 32.
// Double-buffered: async A-panel copy to LDS + early B-panel global loads
// overlap the WMMA work; one barrier per K-step.
// ---------------------------------------------------------------------------
__global__ __launch_bounds__(256) void k_gemm(const bf16_t* __restrict__ A,
                                              const bf16_t* __restrict__ W,
                                              int Ncols, int mode, int rowXor,
                                              bf16_t* __restrict__ outB,
                                              float* __restrict__ outF) {
  __shared__ bf16_t aT[2][128 * 40];   // +8 halves pad -> conflict-free frag reads
  __shared__ bf16_t bT[2][128 * 40];   // stores B^T: [col][k]
  const int t = threadIdx.x;
  const int wid = t >> 5, lane = t & 31;
  const int wm = wid >> 2, wn = wid & 3;       // wave tile: 64 x 32
  const int nn = lane & 15, g = lane >> 4;
  const int mBase = blockIdx.x * 128;
  const int nBase = blockIdx.y * 128;

  v8f acc[4][2];
#pragma unroll
  for (int i = 0; i < 4; ++i)
#pragma unroll
    for (int j = 0; j < 2; ++j) acc[i][j] = zero8();

  const int arowS = t >> 1, aseg = t & 1;       // A staging: 128 rows x 2 segs
  const int bk = t >> 3, bnc = (t & 7) * 16;    // B staging: 32 k x 8 col-chunks
  const size_t asrcRow = (size_t)((mBase + arowS) ^ rowXor);
  const bf16_t* aSrc = A + asrcRow * 512 + aseg * 16;          // + k0
  const bf16_t* bSrc = W + (size_t)bk * Ncols + nBase + bnc;   // + k0*Ncols
  const int aOff = arowS * 40 + aseg * 16;

  APanel ar;
  BPanel br;

  // ---- prologue: stage K-panel 0 into buffer 0 ----
  a_issue(aSrc, &aT[0][aOff], ar);
  b_load(bSrc, br);
  a_commit(&aT[0][aOff], ar);
  b_store(&bT[0][0], bnc, bk, br);
  wait_async();
  __syncthreads();

  for (int i = 0; i < 16; ++i) {
    const int cur = i & 1, nxt = cur ^ 1;
    const int k1 = (i + 1) * 32;
    if (i < 15) {  // issue next panel: async A copy + B loads overlap compute
      a_issue(aSrc + k1, &aT[nxt][aOff], ar);
      b_load(bSrc + (size_t)k1 * Ncols, br);
      if (i < 14)  // global_prefetch_b8 two panels ahead for W
        __builtin_prefetch(bSrc + (size_t)(k1 + 32) * Ncols, 0, 3);
    }

    v16bf aF[4], bF[2];
#pragma unroll
    for (int mt = 0; mt < 4; ++mt)
      aF[mt] = ld_a16(&aT[cur][(wm * 64 + mt * 16 + nn) * 40], g);
#pragma unroll
    for (int nt = 0; nt < 2; ++nt)
      bF[nt] = ld_b16(&bT[cur][(wn * 32 + nt * 16 + nn) * 40], g);
#pragma unroll
    for (int mt = 0; mt < 4; ++mt)
#pragma unroll
      for (int nt = 0; nt < 2; ++nt)
        acc[mt][nt] = WMMA_B(aF[mt], bF[nt], acc[mt][nt]);

    if (i < 15) {  // commit next panel after compute
      a_commit(&aT[nxt][aOff], ar);
      b_store(&bT[nxt][0], bnc, bk, br);
      wait_async();
    }
    __syncthreads();
  }

  // epilogue: C layout -> m = i + 8g, n = lane&15 per tile
#pragma unroll
  for (int mt = 0; mt < 4; ++mt)
#pragma unroll
    for (int nt = 0; nt < 2; ++nt)
#pragma unroll
      for (int i = 0; i < 8; ++i) {
        const int r = mBase + wm * 64 + mt * 16 + i + 8 * g;
        const int c = nBase + wn * 32 + nt * 16 + nn;
        const float v = acc[mt][nt][i];
        if (mode == 2) {
          outF[(size_t)r * 512 + c] = v;
        } else {
          const int bb = r >> 14, n = r & 16383;
          if (c < 512) {
            const int hh = c >> 6, d = c & 63;
            outB[(((size_t)(bb * 8 + hh)) * NSEQ + n) * 64 + d] = (bf16_t)v;
          } else {
            const int c2 = c - 512, hh = c2 >> 6, d = c2 & 63;
            outF[(((size_t)(bb * 8 + hh)) * NSEQ + n) * 64 + d] = v;
          }
        }
      }
}

// ---------------------------------------------------------------------------
// Kernel 4: windowed attention, one wave per (window, head).
// RoPE fused into the Q (A-operand) and K (B-operand) register fragment loads
// (pairs are adjacent halves within a lane in the CDNA5 16-bit layouts).
// LDS/wave (32KB): V^T bf16 [64][96] + score buffer fp32 [64][80] (P overlaid).
// ---------------------------------------------------------------------------
__device__ __forceinline__ v16bf ld_k_frag(const float* tmk, const bf16_t* krow0,
                                           int key, int n0, int ks, int g) {
  F16U u;
  const int dbase = ks * 32 + 8 * g;      // dh = dbase + t + (t & 8)
  if (key < PMTOK) {                       // persistent-memory token, no rope
    const float* r = tmk + key * DH + dbase;
#pragma unroll
    for (int t = 0; t < 16; ++t) u.h[t] = (bf16_t)r[t + (t & 8)];
  } else if (key < PMTOK + SEG) {
    const bf16_t* r = krow0 + (size_t)(key - PMTOK) * DH + dbase;
    const float pos = (float)(n0 + key - PMTOK);
#pragma unroll
    for (int t = 0; t < 16; t += 2) {
      const float x0 = (float)r[t + (t & 8)], x1 = (float)r[t + (t & 8) + 1];
      const float fr = pos * __expf(-LN1E4_OVER_32 * (float)((dbase + t + (t & 8)) >> 1));
      float sn, cs;
      __sincosf(fr, &sn, &cs);
      u.h[t]     = (bf16_t)(x0 * cs - x1 * sn);
      u.h[t + 1] = (bf16_t)(x1 * cs + x0 * sn);
    }
  } else {                                 // pad keys 68..79
#pragma unroll
    for (int t = 0; t < 16; ++t) u.h[t] = (bf16_t)0.f;
  }
  return u.v;
}

__global__ __launch_bounds__(32) void k_attn(const bf16_t* __restrict__ Qb,
                                             const bf16_t* __restrict__ Kb,
                                             const float* __restrict__ Vf,  // orig_v in d_out
                                             const float* __restrict__ TM,
                                             bf16_t* __restrict__ AO) {
  extern __shared__ char smem[];
  bf16_t* vT = (bf16_t*)smem;                 // [64][96] bf16 (V transposed, keys padded)
  float*  sB = (float*)(smem + 12288);        // [64][80] fp32 scores; P bf16 overlaid per row

  const int lane = threadIdx.x;
  const int h    = blockIdx.x & 7;
  const int widx = blockIdx.x >> 3;
  const int b = widx >> 8, win = widx & 255;
  const int n0 = win << 6;
  const size_t hb = ((size_t)(b * 8 + h)) << 14;      // (b*8+h)*NSEQ
  const bf16_t* qrow0 = Qb + (hb + n0) * DH;
  const bf16_t* krow0 = Kb + (hb + n0) * DH;
  const float*  vrow0 = Vf + (hb + n0) * DH;
  const float*  tmk = TM + (size_t)h * PMTOK * DH;
  const float*  tmv = TM + (size_t)HEADS * PMTOK * DH + (size_t)h * PMTOK * DH;

  // ---- stage V^T (keys 0..3 = pm tokens, 4..67 = window, 68..95 = zero pad) ----
  for (int e = lane; e < 96 * 64; e += 32) {
    const int j = e >> 6, d = e & 63;       // d consecutive across lanes -> coalesced reads
    float v;
    if (j < PMTOK)            v = tmv[j * DH + d];
    else if (j < PMTOK + SEG) v = vrow0[(size_t)(j - PMTOK) * DH + d];
    else                      v = 0.f;
    vT[d * 96 + j] = (bf16_t)v;
  }
  asm volatile("" ::: "memory");

  const int rr = lane & 15, g = lane >> 4;

  // ---- Q fragments with rope (kept in registers for all 4 M-tiles) ----
  v16bf aF[4][2];
#pragma unroll
  for (int mt = 0; mt < 4; ++mt) {
    const int m = mt * 16 + rr;
    const bf16_t* qr = qrow0 + (size_t)m * DH;
    const float pos = (float)(n0 + m);
#pragma unroll
    for (int ks = 0; ks < 2; ++ks) {
      F16U u;
#pragma unroll
      for (int c = 0; c < 2; ++c) {
        const int dbase = ks * 32 + c * 16 + 8 * g;
#pragma unroll
        for (int t = 0; t < 8; t += 2) {
          const int d = dbase + t;
          const float x0 = (float)qr[d], x1 = (float)qr[d + 1];
          const float fr = pos * __expf(-LN1E4_OVER_32 * (float)(d >> 1));
          float sn, cs;
          __sincosf(fr, &sn, &cs);
          u.h[c * 8 + t]     = (bf16_t)(x0 * cs - x1 * sn);
          u.h[c * 8 + t + 1] = (bf16_t)(x1 * cs + x0 * sn);
        }
      }
      aF[mt][ks] = u.v;
    }
  }

  // ---- S = (Q K^T) * dh^-0.5, K fragments from global with fused rope ----
#pragma unroll
  for (int nt = 0; nt < 5; ++nt) {
    const int key = nt * 16 + rr;
    v16bf kf0 = ld_k_frag(tmk, krow0, key, n0, 0, g);
    v16bf kf1 = ld_k_frag(tmk, krow0, key, n0, 1, g);
#pragma unroll
    for (int mt = 0; mt < 4; ++mt) {
      v8f c0 = zero8();
      c0 = WMMA_B(aF[mt][0], kf0, c0);
      c0 = WMMA_B(aF[mt][1], kf1, c0);
#pragma unroll
      for (int i = 0; i < 8; ++i)
        sB[(mt * 16 + i + 8 * g) * 80 + nt * 16 + rr] = c0[i] * 0.125f;
    }
  }
  asm volatile("" ::: "memory");

  // ---- softmax over 68 real keys; write P bf16 overlaid (cols 68..95 zero) ----
#pragma unroll
  for (int t = 0; t < 2; ++t) {
    const int m = lane + t * 32;
    float* srow = sB + m * 80;
    bf16_t* prow = (bf16_t*)srow;           // overlay: writes at 2j trail reads at 4j
    float mx = -1e30f;
    for (int j = 0; j < 68; ++j) mx = fmaxf(mx, srow[j]);
    float sum = 0.f;
    for (int j = 0; j < 68; ++j) sum += __expf(srow[j] - mx);
    const float inv = 1.0f / sum;
    for (int j = 0; j < 68; ++j) prow[j] = (bf16_t)(__expf(srow[j] - mx) * inv);
    for (int j = 68; j < 96; ++j) prow[j] = (bf16_t)0.f;
  }
  asm volatile("" ::: "memory");

  // ---- O = P @ V (K padded to 96 -> 3 WMMA K-steps) ----
#pragma unroll
  for (int mt = 0; mt < 4; ++mt) {
    const bf16_t* pr = (const bf16_t*)(sB + (mt * 16 + rr) * 80);
    v16bf aP[3];
#pragma unroll
    for (int kt = 0; kt < 3; ++kt) aP[kt] = ld_a16(pr + kt * 32, g);
#pragma unroll
    for (int nt = 0; nt < 4; ++nt) {
      const int d = nt * 16 + rr;
      v8f acc = zero8();
#pragma unroll
      for (int kt = 0; kt < 3; ++kt) {
        v16bf bV = ld_b16(vT + d * 96 + kt * 32, g);
        acc = WMMA_B(aP[kt], bV, acc);
      }
#pragma unroll
      for (int i = 0; i < 8; ++i) {
        const size_t row = (size_t)b * NSEQ + n0 + mt * 16 + i + 8 * g;
        AO[row * 512 + h * 64 + nt * 16 + rr] = (bf16_t)acc[i];
      }
    }
  }
}

// ---------------------------------------------------------------------------
// Launch
// ---------------------------------------------------------------------------
extern "C" void kernel_launch(void* const* d_in, const int* in_sizes, int n_in,
                              void* d_out, int out_size, void* d_ws, size_t ws_size,
                              hipStream_t stream) {
  (void)in_sizes; (void)n_in; (void)out_size; (void)ws_size;
  const float* seq    = (const float*)d_in[0];
  const float* norm_w = (const float*)d_in[5];
  const float* Wq     = (const float*)d_in[6];
  const float* Wkv    = (const float*)d_in[7];
  const float* Wo     = (const float*)d_in[8];
  const float* TM     = (const float*)d_in[9];

  float* out   = (float*)d_out;                       // (B, N, 512)
  float* origv = out + (size_t)ROWS * DIMM;           // (B, H, N, 64) fp32

  char* ws = (char*)d_ws;
  bf16_t* XN  = (bf16_t*)ws;                          // 64 MB, reused as attn_out later
  bf16_t* QB  = (bf16_t*)(ws + ((size_t)64 << 20));   // 64 MB
  bf16_t* KB  = (bf16_t*)(ws + ((size_t)128 << 20));  // 64 MB
  bf16_t* WQB = (bf16_t*)(ws + ((size_t)192 << 20));  // 512 KB
  bf16_t* WKVB = WQB + 262144;                        // 1 MB
  bf16_t* WOB  = WKVB + 524288;                       // 512 KB
  bf16_t* AO   = XN;                                  // reuse: xn dead after KV GEMM

  k_rmsnorm<<<ROWS / 8, 256, 0, stream>>>(seq, norm_w, XN);
  k_castw<<<4096, 256, 0, stream>>>(Wq, Wkv, Wo, WQB, WKVB, WOB);
  // Q = xn @ Wq
  k_gemm<<<dim3(ROWS / 128, 4), 256, 0, stream>>>(XN, WQB, 512, 0, 0, QB, nullptr);
  // K,V = xn_rev @ Wkv   (rev = row ^ 0x8000); V fp32 straight into d_out orig_v
  k_gemm<<<dim3(ROWS / 128, 8), 256, 0, stream>>>(XN, WKVB, 1024, 1, 0x8000, KB, origv);
  // attention: 1024 windows x 8 heads, one wave each, 32KB dynamic LDS
  k_attn<<<BATCH * NWIN * HEADS, 32, 32768, stream>>>(QB, KB, origv, TM, AO);
  // out = attn_out @ Wo
  k_gemm<<<dim3(ROWS / 128, 4), 256, 0, stream>>>(AO, WOB, 512, 2, 0, nullptr, out);
}
